// E3AttentionPooling_58033598103890
// MI455X (gfx1250) — matrix operand order repaired and checked by reading, not compiled, not run on hardware
//
#include <hip/hip_runtime.h>
#include <math.h>

typedef __attribute__((ext_vector_type(16))) _Float16 v16h;
typedef __attribute__((ext_vector_type(8)))  float    v8f;

// ---------------------------------------------------------------------------
// Irreps: 128x0e + 64x1o + 32x2e.  MULS={128,64,32}, DEGS={1,3,5}, D=480.
// Workspace layout (bytes):
//   [0, 129024)              f16 weights, 3 regions of 21504 halves each:
//                              [ WQ | WC=Wk*Wd^T | WV ]
//                            per region, per-irrep bases (halves): {0,16384,20480}
//   [129024, 129024+4N)      w[N]  (exp(logits))
//   then norm[G], num[G*480] (f32, zeroed each launch)
// Weight storage is in the WMMA B-fragment per-lane layout:
//   index = region*21504 + jbase + ((c*NK+ks)*32 + lane)*16 + e
//   where lane e holds B element (K = ks*32 + 16*(lane>>4) + e, Ncol = c*16 + (lane&15))
// ---------------------------------------------------------------------------

#define WQ_REGION 0
#define WC_REGION 21504
#define WV_REGION 43008
#define W_TOTAL_HALVES 64512

// ---------------- weight prep: pack f16 B-fragments, build Wc = Wk*Wd^T -----
__global__ void prep_weights(
    const float* __restrict__ Wq0, const float* __restrict__ Wq1, const float* __restrict__ Wq2,
    const float* __restrict__ Wk0, const float* __restrict__ Wk1, const float* __restrict__ Wk2,
    const float* __restrict__ Wv0, const float* __restrict__ Wv1, const float* __restrict__ Wv2,
    const float* __restrict__ Wd0, const float* __restrict__ Wd1, const float* __restrict__ Wd2,
    _Float16* __restrict__ wsH)
{
  int idx = blockIdx.x * blockDim.x + threadIdx.x;
  if (idx >= W_TOTAL_HALVES) return;
  int mat = idx / 21504;            // 0=Q, 1=C, 2=V
  int s   = idx - mat * 21504;
  int j, m, jbase;
  if (s < 16384)      { j = 0; m = 128; jbase = 0; }
  else if (s < 20480) { j = 1; m = 64;  jbase = 16384; }
  else                { j = 2; m = 32;  jbase = 20480; }
  int sidx  = s - jbase;
  int chunk = sidx >> 9;            // 512 halves per 32x16 B fragment
  int lane  = (sidx >> 4) & 31;
  int e     = sidx & 15;
  int nk    = m >> 5;
  int c     = chunk / nk;
  int ks    = chunk - c * nk;
  int h     = lane >> 4, r = lane & 15;
  int in_ch  = ks * 32 + h * 16 + e;
  int out_ch = c * 16 + r;

  const float* Wq = (j == 0) ? Wq0 : ((j == 1) ? Wq1 : Wq2);
  const float* Wk = (j == 0) ? Wk0 : ((j == 1) ? Wk1 : Wk2);
  const float* Wv = (j == 0) ? Wv0 : ((j == 1) ? Wv1 : Wv2);
  const float* Wd = (j == 0) ? Wd0 : ((j == 1) ? Wd1 : Wd2);

  float val;
  if (mat == 0) {
    val = Wq[in_ch * m + out_ch];
  } else if (mat == 2) {
    val = Wv[in_ch * m + out_ch];                    // 1/sqrt(m) applied later in f32
  } else {
    float acc = 0.f;                                 // Wc[m_in,i] = sum_t Wk[m_in,t]*Wd[i,t]
    for (int t = 0; t < m; ++t) acc += Wk[in_ch * m + t] * Wd[out_ch * m + t];
    val = acc;
  }
  wsH[idx] = (_Float16)val;
}

// ---------------- A-fragment load: 16 nodes x 32 channels, f32 -> f16 -------
template<int M, int DEG, int FOFF>
__device__ __forceinline__ void load_afrags(const float* __restrict__ f, int row,
                                            int h, int dd, v16h* afr)
{
  const int NK = M / 32;
#pragma unroll
  for (int ks = 0; ks < NK; ++ks) {
#pragma unroll
    for (int e = 0; e < 16; ++e) {
      int K = h * 16 + e;                            // contiguous k order, matches B packing
      afr[ks][e] = (_Float16)f[(long)row * 480 + FOFF + (ks * 32 + K) * DEG + dd];
    }
  }
}

// ---------------- logits contribution of one irrep block --------------------
template<int M, int DEG, int FOFF, int WOFF>
__device__ __forceinline__ void block_logits(const float* __restrict__ f,
    const _Float16* __restrict__ wsH, int row, int lane, float* logit)
{
  const int NK = M / 32, NC = M / 16;
  int h = lane >> 4;
  float dot[8];
#pragma unroll
  for (int v = 0; v < 8; ++v) dot[v] = 0.f;

#pragma unroll
  for (int dd = 0; dd < DEG; ++dd) {
    v16h afr[NK];
    load_afrags<M, DEG, FOFF>(f, row, h, dd, afr);
#pragma unroll 2
    for (int c = 0; c < NC; ++c) {
      v8f qc = {}; v8f kc = {};
#pragma unroll
      for (int ks = 0; ks < NK; ++ks) {
        const _Float16* bq = wsH + WQ_REGION + WOFF + ((size_t)(c * NK + ks) * 32 + lane) * 16;
        v16h bqv = *(const v16h*)bq;
        v16h bcv = *(const v16h*)(bq + WC_REGION);   // same internal offset in Wc region
        qc = __builtin_amdgcn_wmma_f32_16x16x32_f16(false, afr[ks], false, bqv,
                                                    (short)0, qc, false, false);
        kc = __builtin_amdgcn_wmma_f32_16x16x32_f16(false, afr[ks], false, bcv,
                                                    (short)0, kc, false, false);
      }
#pragma unroll
      for (int v = 0; v < 8; ++v) dot[v] += qc[v] * kc[v];
    }
  }
  // full normalization: 1/(m^2 * sqrt(d)) * 1/sqrt(3) * 1/sqrt(480) = 1/(m^2*sqrt(d*1440))
  const float scale = 1.0f / ((float)(M * M) * sqrtf((float)DEG * 1440.0f));
#pragma unroll
  for (int v = 0; v < 8; ++v) logit[v] += scale * dot[v];
}

__global__ __launch_bounds__(256, 4) void logits_kernel(const float* __restrict__ f,
    const int* __restrict__ batch, const _Float16* __restrict__ wsH,
    float* __restrict__ wArr, float* __restrict__ norm, int N)
{
  int wave = threadIdx.x >> 5;
  int lane = threadIdx.x & 31;
  long base = ((long)blockIdx.x * 8 + wave) * 16;
  if (base >= N) return;
  int h = lane >> 4, r = lane & 15;
  int row = (int)(base + r); if (row >= N) row = N - 1;

  // prefetch next tile's rows into cache hierarchy (global_prefetch_b8)
  {
    long pbase = base + 128;                         // next tile handled by this wave slot
    if (pbase + 15 < N) {
      const float* p = f + (pbase + r) * 480 + h * 240;
      __builtin_prefetch(p, 0, 1);
      __builtin_prefetch(p + 60, 0, 1);
      __builtin_prefetch(p + 120, 0, 1);
      __builtin_prefetch(p + 180, 0, 1);
    }
  }

  float logit[8];
#pragma unroll
  for (int v = 0; v < 8; ++v) logit[v] = 0.f;

  block_logits<128, 1, 0,   0>    (f, wsH, row, lane, logit);
  block_logits<64,  3, 128, 16384>(f, wsH, row, lane, logit);
  block_logits<32,  5, 320, 20480>(f, wsH, row, lane, logit);

  // C-fragment: VGPR v, lanes 0-15 -> node base+v, lanes 16-31 -> node base+v+8.
  // Reduce over the 16 channels held across the 16-lane half.
#pragma unroll
  for (int v = 0; v < 8; ++v) {
    float t = logit[v];
    t += __shfl_xor(t, 1, 32);
    t += __shfl_xor(t, 2, 32);
    t += __shfl_xor(t, 4, 32);
    t += __shfl_xor(t, 8, 32);
    if (r == 0) {
      long node = base + v + 8 * h;
      if (node < N) {
        float e = expf(t);
        wArr[node] = e;
        atomicAdd(&norm[batch[node]], e);
      }
    }
  }
}

// ---------------- value + weighted pooling ----------------------------------
template<int M, int DEG, int FOFF, int WOFF>
__device__ __forceinline__ void block_pool(const float* __restrict__ f,
    const _Float16* __restrict__ wsH, const float* wv8, const int* gid8,
    int row, int lane, bool uniform, int g0, float* __restrict__ num,
    long base, int N)
{
  const int NK = M / 32, NC = M / 16;
  int h = lane >> 4, r = lane & 15;
  const float vscale = 1.0f / sqrtf((float)M);
#pragma unroll
  for (int dd = 0; dd < DEG; ++dd) {
    v16h afr[NK];
    load_afrags<M, DEG, FOFF>(f, row, h, dd, afr);
#pragma unroll 2
    for (int c = 0; c < NC; ++c) {
      v8f vc = {};
#pragma unroll
      for (int ks = 0; ks < NK; ++ks) {
        const _Float16* bv = wsH + WV_REGION + WOFF + ((size_t)(c * NK + ks) * 32 + lane) * 16;
        vc = __builtin_amdgcn_wmma_f32_16x16x32_f16(false, afr[ks], false, *(const v16h*)bv,
                                                    (short)0, vc, false, false);
      }
      int chan = FOFF + (c * 16 + r) * DEG + dd;
      if (uniform) {
        // all 16 nodes in one graph (batch sorted): reduce in-register first
        float acc = 0.f;
#pragma unroll
        for (int v = 0; v < 8; ++v) acc += vc[v] * wv8[v];
        acc *= vscale;
        acc += __shfl_xor(acc, 16, 32);              // combine node halves (same channel)
        if (h == 0) atomicAdd(&num[(size_t)g0 * 480 + chan], acc);
      } else {
#pragma unroll
        for (int v = 0; v < 8; ++v) {
          long node = base + v + 8 * h;
          if (node < N)
            atomicAdd(&num[(size_t)gid8[v] * 480 + chan], vc[v] * wv8[v] * vscale);
        }
      }
    }
  }
}

__global__ __launch_bounds__(256, 4) void pool_kernel(const float* __restrict__ f,
    const int* __restrict__ batch, const _Float16* __restrict__ wsH,
    const float* __restrict__ wArr, float* __restrict__ num, int N)
{
  int wave = threadIdx.x >> 5;
  int lane = threadIdx.x & 31;
  long base = ((long)blockIdx.x * 8 + wave) * 16;
  if (base >= N) return;
  int h = lane >> 4, r = lane & 15;
  int row = (int)(base + r); if (row >= N) row = N - 1;

  {
    long pbase = base + 128;
    if (pbase + 15 < N) {
      const float* p = f + (pbase + r) * 480 + h * 240;
      __builtin_prefetch(p, 0, 1);
      __builtin_prefetch(p + 60, 0, 1);
      __builtin_prefetch(p + 120, 0, 1);
      __builtin_prefetch(p + 180, 0, 1);
    }
  }

  float wv8[8]; int gid8[8];
#pragma unroll
  for (int v = 0; v < 8; ++v) {
    long node = base + v + 8 * h;
    if (node < N) { wv8[v] = wArr[node]; gid8[v] = batch[node]; }
    else          { wv8[v] = 0.f;        gid8[v] = 0; }
  }
  long last = base + 15; if (last >= N) last = N - 1;
  int g0 = batch[base];
  bool uniform = (g0 == batch[last]);

  block_pool<128, 1, 0,   0>    (f, wsH, wv8, gid8, row, lane, uniform, g0, num, base, N);
  block_pool<64,  3, 128, 16384>(f, wsH, wv8, gid8, row, lane, uniform, g0, num, base, N);
  block_pool<32,  5, 320, 20480>(f, wsH, wv8, gid8, row, lane, uniform, g0, num, base, N);
}

// ---------------- finalize: out = num / clip(norm) --------------------------
__global__ void finalize_kernel(const float* __restrict__ num,
                                const float* __restrict__ norm,
                                float* __restrict__ out, int total)
{
  int i = blockIdx.x * blockDim.x + threadIdx.x;
  if (i >= total) return;
  float nz = norm[i / 480];
  nz = fmaxf(nz, 1e-8f);
  out[i] = num[i] / nz;
}

// ---------------------------------------------------------------------------
extern "C" void kernel_launch(void* const* d_in, const int* in_sizes, int n_in,
                              void* d_out, int out_size, void* d_ws, size_t ws_size,
                              hipStream_t stream)
{
  const float* f  = (const float*)d_in[0];
  const float* Wq0 = (const float*)d_in[1],  *Wq1 = (const float*)d_in[2],  *Wq2 = (const float*)d_in[3];
  const float* Wk0 = (const float*)d_in[4],  *Wk1 = (const float*)d_in[5],  *Wk2 = (const float*)d_in[6];
  const float* Wv0 = (const float*)d_in[7],  *Wv1 = (const float*)d_in[8],  *Wv2 = (const float*)d_in[9];
  const float* Wd0 = (const float*)d_in[10], *Wd1 = (const float*)d_in[11], *Wd2 = (const float*)d_in[12];
  const int*   batch = (const int*)d_in[13];

  int N = in_sizes[0] / 480;
  int G = out_size / 480;

  char* ws = (char*)d_ws;
  _Float16* wsH = (_Float16*)ws;
  float* wArr = (float*)(ws + W_TOTAL_HALVES * sizeof(_Float16)); // 129024B
  float* norm = wArr + N;
  float* num  = norm + G;

  // zero accumulators (norm + num contiguous)
  hipMemsetAsync(norm, 0, (size_t)(G + (size_t)G * 480) * sizeof(float), stream);

  prep_weights<<<(W_TOTAL_HALVES + 255) / 256, 256, 0, stream>>>(
      Wq0, Wq1, Wq2, Wk0, Wk1, Wk2, Wv0, Wv1, Wv2, Wd0, Wd1, Wd2, wsH);

  int tiles  = (N + 15) / 16;
  int blocks = (tiles + 7) / 8;        // 8 waves per block, one 16-node tile per wave

  logits_kernel<<<blocks, 256, 0, stream>>>(f, batch, wsH, wArr, norm, N);
  pool_kernel  <<<blocks, 256, 0, stream>>>(f, batch, wsH, wArr, num, N);
  finalize_kernel<<<(G * 480 + 255) / 256, 256, 0, stream>>>(num, norm, (float*)d_out, G * 480);
}
